// FiLMBlock_71390946394386
// MI455X (gfx1250) — matrix operand order repaired and checked by reading, not compiled, not run on hardware
//
#include <hip/hip_runtime.h>
#include <math.h>

typedef __attribute__((ext_vector_type(8)))  _Float16 v8h;
typedef __attribute__((ext_vector_type(16))) _Float16 v16h;
typedef __attribute__((ext_vector_type(8)))  float    v8f;

// ---------------------------------------------------------------------------
// WMMA fragment loaders for v_wmma_f32_16x16x32_f16 (CDNA5 wave32 layouts).
// A (16x32, M x K), row-major with leading dim lda (in halves):
//   lanes 0-15  hold row M=lane,   K = 0..7  and 16..23
//   lanes 16-31 hold row M=lane-16,K = 8..15 and 24..31
// B (32x16, K x N): column n is stored as row n of a (N x K) weight matrix:
//   lanes 0-15  hold col N=lane,   K = 0..7  and 16..23
//   lanes 16-31 hold col N=lane-16,K = 8..15 and 24..31
// ---------------------------------------------------------------------------
__device__ __forceinline__ v16h load_a_frag(const _Float16* Ak, int lda) {
  const int lane = threadIdx.x & 31;
  const _Float16* p = Ak + (size_t)(lane & 15) * lda + ((lane >> 4) << 3);
  v8h lo = *(const v8h*)p;
  v8h hi = *(const v8h*)(p + 16);
  v16h r;
#pragma unroll
  for (int i = 0; i < 8; ++i) { r[i] = lo[i]; r[i + 8] = hi[i]; }
  return r;
}

__device__ __forceinline__ v16h load_b_frag(const _Float16* Wk, int ldb) {
  const int lane = threadIdx.x & 31;
  const _Float16* p = Wk + (size_t)(lane & 15) * ldb + ((lane >> 4) << 3);
  v8h lo = *(const v8h*)p;
  v8h hi = *(const v8h*)(p + 16);
  v16h r;
#pragma unroll
  for (int i = 0; i < 8; ++i) { r[i] = lo[i]; r[i + 8] = hi[i]; }
  return r;
}

#define WMMA_F16(a, b, c) \
  __builtin_amdgcn_wmma_f32_16x16x32_f16(false, (a), false, (b), (short)0, (c), false, false)

// Single-tile (16x16) K-loop: used by the attention kernel.
__device__ __forceinline__ v8f wmma_kloop(const _Float16* A, int lda,
                                          const _Float16* W, int ldb,
                                          int K, v8f acc) {
  for (int k = 0; k < K; k += 32) {
    v16h a = load_a_frag(A + k, lda);
    v16h b = load_b_frag(W + k, ldb);
    acc = WMMA_F16(a, b, acc);
  }
  return acc;
}

// 2x2 register-blocked K-loop: wave computes a 32x32 tile (4 accumulators).
// Each k-step: 2 A frags + 2 B frags feed 4 WMMAs (halved operand traffic
// per WMMA vs single-tile).
__device__ __forceinline__ void wmma_kloop2x2(const _Float16* A, int lda,
                                              const _Float16* W, int ldb,
                                              int K, v8f acc[4]) {
  for (int k = 0; k < K; k += 32) {
    v16h a0 = load_a_frag(A + k, lda);
    v16h a1 = load_a_frag(A + (size_t)16 * lda + k, lda);
    v16h b0 = load_b_frag(W + k, ldb);
    v16h b1 = load_b_frag(W + (size_t)16 * ldb + k, ldb);
    acc[0] = WMMA_F16(a0, b0, acc[0]);
    acc[1] = WMMA_F16(a0, b1, acc[1]);
    acc[2] = WMMA_F16(a1, b0, acc[2]);
    acc[3] = WMMA_F16(a1, b1, acc[3]);
  }
}

// ---------------------------------------------------------------------------
// K0: FiLM coefficients  coeff[which][b][d] = t[b] . G[d]   (4 x 8 x 256)
// ---------------------------------------------------------------------------
__global__ void k_coeffs(const float* __restrict__ t,
                         const float* __restrict__ an_g,
                         const float* __restrict__ an_b,
                         const float* __restrict__ fn_g,
                         const float* __restrict__ fn_b,
                         float* __restrict__ coeff) {
  const int which = blockIdx.x;   // 0:ga 1:ba 2:gf 3:bf
  const int b = blockIdx.y;
  const int d = threadIdx.x;
  const float* G = (which == 0) ? an_g : (which == 1) ? an_b
                 : (which == 2) ? fn_g : fn_b;
  const float* tv = t + b * 256;
  const float* gr = G + d * 256;
  float s = 0.f;
  for (int k = 0; k < 256; ++k) s += tv[k] * gr[k];
  coeff[which * 2048 + b * 256 + d] = s;
}

// K0b: f32 -> f16 weight conversion
__global__ void k_f2h(const float* __restrict__ src, _Float16* __restrict__ dst,
                      int n) {
  int i = blockIdx.x * 256 + threadIdx.x;
  if (i < n) dst[i] = (_Float16)src[i];
}

// ---------------------------------------------------------------------------
// K1: RMS-FiLM over channels + tokenize into window order (f16).
// One block = one pixel, 256 threads = channels.
// Window token id: ((b*16+wh)*16+ww)*64 + (ih*8+iw)
// ---------------------------------------------------------------------------
__global__ void k_film_attn_in(const float* __restrict__ x,
                               const float* __restrict__ coeff,
                               _Float16* __restrict__ A0) {
  __shared__ float red[256];
  const int pix = blockIdx.x;
  const int b = pix >> 14;
  const int hw = pix & 16383;
  const int c = threadIdx.x;
  const float xv = x[(((size_t)b * 256 + c) << 14) + hw];
  red[c] = xv * xv;
  __syncthreads();
  for (int s = 128; s > 0; s >>= 1) {
    if (c < s) red[c] += red[c + s];
    __syncthreads();
  }
  const float scale = rsqrtf(red[0] * (1.f / 256.f) + 1e-6f);
  const float y = coeff[b * 256 + c] * xv * scale + coeff[2048 + b * 256 + c];
  const int h = hw >> 7, w = hw & 127;
  const int tok = (((b * 16 + (h >> 3)) * 16 + (w >> 3)) << 6) +
                  ((h & 7) * 8 + (w & 7));
  A0[(size_t)tok * 256 + c] = (_Float16)y;
}

// ---------------------------------------------------------------------------
// K2: QKV projections.  grid = (2048, 4, 3), block = 128 (4 waves as 2x2).
// Block computes a 64(token) x 64(outchan) tile; wave computes 32x32.
// z=2 (V) stores transposed per (window, head): v[win][head][d][l].
// ---------------------------------------------------------------------------
__global__ void k_gemm_qkv(const _Float16* __restrict__ A0,
                           const _Float16* __restrict__ wq16,
                           const _Float16* __restrict__ wk16,
                           const _Float16* __restrict__ wv16,
                           _Float16* __restrict__ q16,
                           _Float16* __restrict__ k16,
                           _Float16* __restrict__ v16b) {
  __shared__ _Float16 Ws[64 * 256];  // 32 KB weight slab (64 outchan x K)
  const int tid = threadIdx.x;
  const int z = blockIdx.z;
  const _Float16* Wt = (z == 0) ? wq16 : (z == 1) ? wk16 : wv16;
  const int nB0 = blockIdx.y << 6;
  {
    const v8h* src = (const v8h*)(Wt + (size_t)nB0 * 256);
    v8h* dst = (v8h*)Ws;
#pragma unroll
    for (int i = tid; i < 64 * 256 / 8; i += 128) dst[i] = src[i];
  }
  const int wave = tid >> 5;
  const int mBase = (blockIdx.x << 6) + ((wave & 1) << 5);
  const int nOff = (wave >> 1) << 5;  // 0 or 32 within slab
  __builtin_prefetch(A0 + (size_t)mBase * 256, 0, 1);  // global_prefetch_b8
  __syncthreads();
  v8f acc[4] = {};
  wmma_kloop2x2(A0 + (size_t)mBase * 256, 256, Ws + (size_t)nOff * 256, 256,
                256, acc);
  const int lane = tid & 31;
  const int hi = lane >> 4, nc = lane & 15;
#pragma unroll
  for (int t2 = 0; t2 < 4; ++t2) {
    const int mT = mBase + ((t2 >> 1) << 4);
    const int nT = nB0 + nOff + ((t2 & 1) << 4);
#pragma unroll
    for (int r = 0; r < 8; ++r) {
      const int m = mT + r + hi * 8;
      const int n = nT + nc;
      const _Float16 hv = (_Float16)acc[t2][r];
      if (z == 2) {
        const size_t addr = (size_t)(m >> 6) * 16384 + (size_t)(n >> 5) * 2048 +
                            (size_t)(n & 31) * 64 + (m & 63);
        v16b[addr] = hv;
      } else {
        _Float16* out = (z == 0) ? q16 : k16;
        out[(size_t)m * 256 + n] = hv;
      }
    }
  }
}

// ---------------------------------------------------------------------------
// K3: window attention.  grid = (2048 windows, 8 heads), block = 128.
// S = q k^T (WMMA, K=32) -> LDS f32; softmax(+rel-pos bias) -> LDS f16;
// O = P v (WMMA, K=64) -> O_buf token-major f16.
// ---------------------------------------------------------------------------
__global__ void k_attn(const _Float16* __restrict__ q16,
                       const _Float16* __restrict__ k16,
                       const _Float16* __restrict__ v16b,
                       const float* __restrict__ rpb,
                       _Float16* __restrict__ Ob) {
  __shared__ float    Sb[64 * 64];
  __shared__ _Float16 Pb[64 * 64];
  const int tid = threadIdx.x;
  const int lane = tid & 31;
  const int wave = tid >> 5;
  const int win = blockIdx.x;
  const int head = blockIdx.y;
  const _Float16* qb  = q16  + (size_t)win * 64 * 256 + head * 32;
  const _Float16* kb  = k16  + (size_t)win * 64 * 256 + head * 32;
  const _Float16* vtb = v16b + (size_t)win * 16384 + head * 2048;
  const int mBase = wave << 4;
  const int hi = lane >> 4, nc = lane & 15;

#pragma unroll
  for (int nb = 0; nb < 4; ++nb) {
    v8f acc = {};
    acc = wmma_kloop(qb + (size_t)mBase * 256, 256,
                     kb + (size_t)(nb << 4) * 256, 256, 32, acc);
#pragma unroll
    for (int r = 0; r < 8; ++r)
      Sb[(mBase + r + hi * 8) * 64 + (nb << 4) + nc] = acc[r];
  }
  __syncthreads();

  if (tid < 64) {
    const int l = tid;
    const float sc = 0.17677669529663687f;  // 1/sqrt(32)
    const float* rb = rpb + head * 225;
    const int l0 = l >> 3, l1 = l & 7;
    float mx = -1e30f;
    for (int m = 0; m < 64; ++m) {
      const int idx = (l0 - (m >> 3) + 7) * 15 + (l1 - (m & 7) + 7);
      const float s = Sb[l * 64 + m] * sc + rb[idx];
      Sb[l * 64 + m] = s;
      mx = fmaxf(mx, s);
    }
    float sum = 0.f;
    for (int m = 0; m < 64; ++m) {
      const float e = __expf(Sb[l * 64 + m] - mx);
      Sb[l * 64 + m] = e;
      sum += e;
    }
    const float inv = 1.f / sum;
    for (int m = 0; m < 64; ++m)
      Pb[l * 64 + m] = (_Float16)(Sb[l * 64 + m] * inv);
  }
  __syncthreads();

#pragma unroll
  for (int nb = 0; nb < 2; ++nb) {
    v8f acc = {};
    acc = wmma_kloop(Pb + mBase * 64, 64, vtb + (nb << 4) * 64, 64, 64, acc);
#pragma unroll
    for (int r = 0; r < 8; ++r) {
      const int m = mBase + r + hi * 8;
      const int n = (nb << 4) + nc;
      Ob[(size_t)(win * 64 + m) * 256 + head * 32 + n] = (_Float16)acc[r];
    }
  }
}

// ---------------------------------------------------------------------------
// K4: output projection + residual, de-tokenize window order -> NCHW f32.
// grid = (2048, 4), block = 128 (2x2 waves, 64x64 block tile).
// ---------------------------------------------------------------------------
__global__ void k_gemm_oproj(const _Float16* __restrict__ Ob,
                             const _Float16* __restrict__ wo16,
                             const float* __restrict__ wo_b,
                             const float* __restrict__ x0,
                             float* __restrict__ x1) {
  __shared__ _Float16 Ws[64 * 256];
  const int tid = threadIdx.x;
  const int nB0 = blockIdx.y << 6;
  {
    const v8h* src = (const v8h*)(wo16 + (size_t)nB0 * 256);
    v8h* dst = (v8h*)Ws;
#pragma unroll
    for (int i = tid; i < 64 * 256 / 8; i += 128) dst[i] = src[i];
  }
  __syncthreads();
  const int wave = tid >> 5;
  const int mBase = (blockIdx.x << 6) + ((wave & 1) << 5);
  const int nOff = (wave >> 1) << 5;
  v8f acc[4] = {};
  wmma_kloop2x2(Ob + (size_t)mBase * 256, 256, Ws + (size_t)nOff * 256, 256,
                256, acc);
  const int lane = tid & 31;
  const int hi = lane >> 4, nc = lane & 15;
#pragma unroll
  for (int t2 = 0; t2 < 4; ++t2) {
    const int mT = mBase + ((t2 >> 1) << 4);
    const int nT = nB0 + nOff + ((t2 & 1) << 4);
#pragma unroll
    for (int r = 0; r < 8; ++r) {
      const int m = mT + r + hi * 8;
      const int n = nT + nc;
      const int wn = m >> 6, l = m & 63;
      const int b = wn >> 8, wh = (wn >> 4) & 15, ww = wn & 15;
      const int hh = wh * 8 + (l >> 3), wwp = ww * 8 + (l & 7);
      const size_t addr = (((size_t)b * 256 + n) << 14) + (hh << 7) + wwp;
      x1[addr] = acc[t2][r] + wo_b[n] + x0[addr];
    }
  }
}

// ---------------------------------------------------------------------------
// K5: fused depthwise 7x7 conv + RMS-FiLM, tokens in plain (b,h,w) order.
// One block = one pixel, 256 threads = channels.
// ---------------------------------------------------------------------------
__global__ void k_dwconv_film(const float* __restrict__ x1,
                              const float* __restrict__ dw_w,
                              const float* __restrict__ dw_b,
                              const float* __restrict__ coeff,
                              _Float16* __restrict__ A1) {
  __shared__ float red[256];
  const int pix = blockIdx.x;
  const int b = pix >> 14;
  const int hw = pix & 16383;
  const int h = hw >> 7, w = hw & 127;
  const int c = threadIdx.x;
  const float* xc = x1 + (((size_t)b * 256 + c) << 14);
  const float* wc = dw_w + c * 49;
  float acc = dw_b[c];
  for (int dy = 0; dy < 7; ++dy) {
    const int hh = h + dy - 3;
    if (hh < 0 || hh > 127) continue;
    for (int dx = 0; dx < 7; ++dx) {
      const int ww = w + dx - 3;
      if (ww < 0 || ww > 127) continue;
      acc += xc[(hh << 7) + ww] * wc[dy * 7 + dx];
    }
  }
  red[c] = acc * acc;
  __syncthreads();
  for (int s = 128; s > 0; s >>= 1) {
    if (c < s) red[c] += red[c + s];
    __syncthreads();
  }
  const float scale = rsqrtf(red[0] * (1.f / 256.f) + 1e-6f);
  const float y = coeff[4096 + b * 256 + c] * acc * scale +
                  coeff[6144 + b * 256 + c];
  A1[(size_t)pix * 256 + c] = (_Float16)y;
}

// ---------------------------------------------------------------------------
// K6: MLP up-projection (256 -> 1024) + exact GELU, f16 out.
// grid = (2048, 16), block = 128 (2x2 waves).
// ---------------------------------------------------------------------------
__global__ void k_mlp1(const _Float16* __restrict__ A1,
                       const _Float16* __restrict__ w116,
                       const float* __restrict__ b1,
                       _Float16* __restrict__ H1) {
  __shared__ _Float16 Ws[64 * 256];
  const int tid = threadIdx.x;
  const int nB0 = blockIdx.y << 6;
  {
    const v8h* src = (const v8h*)(w116 + (size_t)nB0 * 256);
    v8h* dst = (v8h*)Ws;
#pragma unroll
    for (int i = tid; i < 64 * 256 / 8; i += 128) dst[i] = src[i];
  }
  __syncthreads();
  const int wave = tid >> 5;
  const int mBase = (blockIdx.x << 6) + ((wave & 1) << 5);
  const int nOff = (wave >> 1) << 5;
  v8f acc[4] = {};
  wmma_kloop2x2(A1 + (size_t)mBase * 256, 256, Ws + (size_t)nOff * 256, 256,
                256, acc);
  const int lane = tid & 31;
  const int hi = lane >> 4, nc = lane & 15;
#pragma unroll
  for (int t2 = 0; t2 < 4; ++t2) {
    const int mT = mBase + ((t2 >> 1) << 4);
    const int nT = nB0 + nOff + ((t2 & 1) << 4);
#pragma unroll
    for (int r = 0; r < 8; ++r) {
      const int m = mT + r + hi * 8;
      const int n = nT + nc;
      const float v = acc[t2][r] + b1[n];
      const float g = 0.5f * v * (1.f + erff(v * 0.70710678118654752f));
      H1[(size_t)m * 1024 + n] = (_Float16)g;
    }
  }
}

// ---------------------------------------------------------------------------
// K7: MLP down-projection (1024 -> 256) + residual -> d_out (NCHW f32).
// grid = (2048, 4), block = 128.  K=1024 staged in four 256-wide LDS chunks.
// ---------------------------------------------------------------------------
__global__ void k_mlp2(const _Float16* __restrict__ H1,
                       const _Float16* __restrict__ w216,
                       const float* __restrict__ b2,
                       const float* __restrict__ x1,
                       float* __restrict__ out) {
  __shared__ _Float16 Ws[64 * 256];  // 32 KB chunk of the weight slab
  const int tid = threadIdx.x;
  const int nB0 = blockIdx.y << 6;
  const int wave = tid >> 5;
  const int mBase = (blockIdx.x << 6) + ((wave & 1) << 5);
  const int nOff = (wave >> 1) << 5;
  v8f acc[4] = {};
  for (int kc = 0; kc < 1024; kc += 256) {
#pragma unroll
    for (int i = tid; i < 64 * 32; i += 128) {  // 64 rows x 32 v8h
      const int row = i >> 5, c8 = (i & 31) << 3;
      *(v8h*)(Ws + row * 256 + c8) =
          *(const v8h*)(w216 + (size_t)(nB0 + row) * 1024 + kc + c8);
    }
    __syncthreads();
    wmma_kloop2x2(H1 + (size_t)mBase * 1024 + kc, 1024,
                  Ws + (size_t)nOff * 256, 256, 256, acc);
    __syncthreads();
  }
  const int lane = tid & 31;
  const int hi = lane >> 4, nc = lane & 15;
#pragma unroll
  for (int t2 = 0; t2 < 4; ++t2) {
    const int mT = mBase + ((t2 >> 1) << 4);
    const int nT = nB0 + nOff + ((t2 & 1) << 4);
#pragma unroll
    for (int r = 0; r < 8; ++r) {
      const int m = mT + r + hi * 8;
      const int n = nT + nc;
      const int b = m >> 14;
      const size_t addr = (((size_t)b * 256 + n) << 14) + (m & 16383);
      out[addr] = acc[t2][r] + b2[n] + x1[addr];
    }
  }
}

// ---------------------------------------------------------------------------
extern "C" void kernel_launch(void* const* d_in, const int* in_sizes, int n_in,
                              void* d_out, int out_size, void* d_ws,
                              size_t ws_size, hipStream_t stream) {
  (void)in_sizes; (void)n_in; (void)out_size; (void)ws_size;
  const float* x    = (const float*)d_in[0];
  const float* t    = (const float*)d_in[1];
  const float* an_g = (const float*)d_in[2];
  const float* an_b = (const float*)d_in[3];
  const float* wq   = (const float*)d_in[4];
  const float* wk   = (const float*)d_in[5];
  const float* wv   = (const float*)d_in[6];
  const float* wo_w = (const float*)d_in[7];
  const float* wo_b = (const float*)d_in[8];
  const float* rpb  = (const float*)d_in[9];
  const float* dw_w = (const float*)d_in[10];
  const float* dw_b = (const float*)d_in[11];
  const float* fn_g = (const float*)d_in[12];
  const float* fn_b = (const float*)d_in[13];
  const float* w1   = (const float*)d_in[14];
  const float* b1   = (const float*)d_in[15];
  const float* w2   = (const float*)d_in[16];
  const float* b2   = (const float*)d_in[17];
  float* out = (float*)d_out;

  // workspace carve-up
  char* w = (char*)d_ws;
  float*    coeff = (float*)w;                       // 4*8*256 f32 = 32 KB
  _Float16* wq16  = (_Float16*)(w + 32768);
  _Float16* wk16  = wq16 + 65536;
  _Float16* wv16  = wk16 + 65536;
  _Float16* wo16  = wv16 + 65536;
  _Float16* w116  = wo16 + 65536;                    // 262144 halves
  _Float16* w216  = w116 + 262144;                   // 262144 halves
  _Float16* A0    = (_Float16*)(w + 1605632);        // 131072*256 halves (64MB)
  _Float16* q16   = A0   + 33554432;
  _Float16* k16   = q16  + 33554432;
  _Float16* v16b  = k16  + 33554432;
  _Float16* Ob    = v16b + 33554432;
  float*    x1    = (float*)(Ob + 33554432);         // 128 MB f32
  _Float16* H1    = q16;  // alias dead q/k/v/O region (exactly 256 MB)
  _Float16* A1    = A0;   // alias dead A0

  // FiLM coefficients + weight casts
  k_coeffs<<<dim3(4, 8), 256, 0, stream>>>(t, an_g, an_b, fn_g, fn_b, coeff);
  k_f2h<<<256, 256, 0, stream>>>(wq, wq16, 65536);
  k_f2h<<<256, 256, 0, stream>>>(wk, wk16, 65536);
  k_f2h<<<256, 256, 0, stream>>>(wv, wv16, 65536);
  k_f2h<<<256, 256, 0, stream>>>(wo_w, wo16, 65536);
  k_f2h<<<1024, 256, 0, stream>>>(w1, w116, 262144);
  k_f2h<<<1024, 256, 0, stream>>>(w2, w216, 262144);

  // Attention branch
  k_film_attn_in<<<131072, 256, 0, stream>>>(x, coeff, A0);
  k_gemm_qkv<<<dim3(2048, 4, 3), 128, 0, stream>>>(A0, wq16, wk16, wv16,
                                                   q16, k16, v16b);
  k_attn<<<dim3(2048, 8), 128, 0, stream>>>(q16, k16, v16b, rpb, Ob);
  k_gemm_oproj<<<dim3(2048, 4), 128, 0, stream>>>(Ob, wo16, wo_b, x, x1);

  // ConvNeXt branch
  k_dwconv_film<<<131072, 256, 0, stream>>>(x1, dw_w, dw_b, coeff, A1);
  k_mlp1<<<dim3(2048, 16), 128, 0, stream>>>(A1, w116, b1, H1);
  k_mlp2<<<dim3(2048, 4), 128, 0, stream>>>(H1, w216, b2, x1, out);
}
